// Model_54202487275578
// MI455X (gfx1250) — compile-verified, matrix operand order checked by reading, not run on hardware
//
#include <hip/hip_runtime.h>
#include <hip/hip_bf16.h>

#define B_    32
#define T_    50
#define NN_   60
#define EE_   512
#define HID_  1024
#define OUTH_ 256
#define BT_   (B_ * T_)

typedef __attribute__((ext_vector_type(16))) __bf16 v16bf;
typedef __attribute__((ext_vector_type(8)))  __bf16 v8bf;
typedef __attribute__((ext_vector_type(8)))  float  v8f;

// ---------------------------------------------------------------------------
// WMMA fragment builder (CDNA5 16-bit A/B layout, wave32)
//   lanes 0-15 : rows/cols 0-15, K in {kk+0..7} U {kk+16..23}
//   lanes 16-31: rows/cols 0-15, K in {kk+8..15} U {kk+24..31}
// Works for global bf16 and LDS bf16 pointers alike (two 16B vector loads).
// ---------------------------------------------------------------------------
__device__ __forceinline__ v16bf frag_bf16(const __bf16* base, int ld,
                                           int row, int kk, int lane) {
  int r  = row + (lane & 15);
  int kh = (lane >> 4) & 1;
  const __bf16* p = base + (size_t)r * ld + kk + kh * 8;
  v8bf lo = *(const v8bf*)(p);
  v8bf hi = *(const v8bf*)(p + 16);
  v16bf f;
#pragma unroll
  for (int i = 0; i < 8; ++i) { f[i] = lo[i]; f[8 + i] = hi[i]; }
  return f;
}

__device__ __forceinline__ v8f wmma_bf16(v16bf a, v16bf b, v8f c) {
  return __builtin_amdgcn_wmma_f32_16x16x32_bf16(false, a, false, b,
                                                 (short)0, c, false, false);
}

__device__ __forceinline__ float sigmf(float x) { return 1.f / (1.f + __expf(-x)); }

// Async global->LDS copy, 16 bytes per lane (CDNA5 GLOBAL_LOAD_ASYNC_TO_LDS_B128)
__device__ __forceinline__ void async_g2lds_b128(unsigned lds_addr, const void* gptr) {
  asm volatile("global_load_async_to_lds_b128 %0, %1, off"
               :: "v"(lds_addr), "v"((unsigned long long)(uintptr_t)gptr)
               : "memory");
}
__device__ __forceinline__ void wait_asynccnt0() {
  asm volatile("s_wait_asynccnt 0x0" ::: "memory");
}

// ---------------------------------------------------------------------------
// Utility kernels
// ---------------------------------------------------------------------------
__global__ void f32_to_bf16_k(const float* __restrict__ s,
                              __hip_bfloat16* __restrict__ d, long long n) {
  long long i = (long long)blockIdx.x * blockDim.x + threadIdx.x;
  if (i < n) d[i] = __float2bfloat16(s[i]);
}

__global__ void zero_f32_k(float* __restrict__ p, int n) {
  int i = blockIdx.x * blockDim.x + threadIdx.x;
  if (i < n) p[i] = 0.f;
}

__global__ void zero_b16_k(__hip_bfloat16* __restrict__ p, int n) {
  int i = blockIdx.x * blockDim.x + threadIdx.x;
  if (i < n) p[i] = __float2bfloat16(0.f);
}

// ---------------------------------------------------------------------------
// GCN: one workgroup per graph; LDS float atomics for degree + segment sums
// ---------------------------------------------------------------------------
__global__ void gcn_kernel(const float* __restrict__ x, const int* __restrict__ ei,
                           const float* __restrict__ W1, const float* __restrict__ b1,
                           const float* __restrict__ W2, const float* __restrict__ b2,
                           float* __restrict__ res) {
  int g = blockIdx.x;                       // 0 .. B*T-1
  const float* xg  = x  + (size_t)g * NN_ * 2;
  const int*   src = ei + (size_t)g * 2 * EE_;
  const int*   dst = src + EE_;

  __shared__ float sxw[NN_ * 4];
  __shared__ float sdeg[NN_];
  __shared__ float sdinv[NN_];
  __shared__ float sh[NN_ * 4];
  __shared__ float sout[2];

  int tid = threadIdx.x;                    // 128 threads
  for (int i = tid; i < NN_ * 4; i += blockDim.x) sh[i] = 0.f;
  for (int i = tid; i < NN_;     i += blockDim.x) sdeg[i] = 0.f;
  if (tid < 2) sout[tid] = 0.f;
  __syncthreads();

  for (int e = tid; e < EE_; e += blockDim.x) atomicAdd(&sdeg[dst[e]], 1.0f);
  for (int i = tid; i < NN_; i += blockDim.x) atomicAdd(&sdeg[i], 1.0f);

  for (int i = tid; i < NN_ * 4; i += blockDim.x) {
    int node = i >> 2, c = i & 3;
    sxw[i] = xg[node * 2 + 0] * W1[0 * 4 + c] + xg[node * 2 + 1] * W1[1 * 4 + c];
  }
  __syncthreads();
  for (int i = tid; i < NN_; i += blockDim.x)
    sdinv[i] = (sdeg[i] > 0.f) ? rsqrtf(sdeg[i]) : 0.f;
  __syncthreads();

  for (int e = tid; e < EE_ + NN_; e += blockDim.x) {
    int s = (e < EE_) ? src[e] : (e - EE_);
    int d = (e < EE_) ? dst[e] : (e - EE_);
    float nm = sdinv[s] * sdinv[d];
#pragma unroll
    for (int c = 0; c < 4; ++c) atomicAdd(&sh[d * 4 + c], sxw[s * 4 + c] * nm);
  }
  __syncthreads();
  for (int i = tid; i < NN_ * 4; i += blockDim.x) sh[i] += b1[i & 3];
  __syncthreads();

  for (int e = tid; e < EE_ + NN_; e += blockDim.x) {
    int s = (e < EE_) ? src[e] : (e - EE_);
    int d = (e < EE_) ? dst[e] : (e - EE_);
    if (d != 0) continue;
    float nm = sdinv[s] * sdinv[d];
    float h0 = sh[s * 4 + 0], h1 = sh[s * 4 + 1], h2 = sh[s * 4 + 2], h3 = sh[s * 4 + 3];
#pragma unroll
    for (int c = 0; c < 2; ++c) {
      float v = h0 * W2[0 * 2 + c] + h1 * W2[1 * 2 + c] +
                h2 * W2[2 * 2 + c] + h3 * W2[3 * 2 + c];
      atomicAdd(&sout[c], v * nm);
    }
  }
  __syncthreads();
  if (tid < 2) res[g * 2 + tid] = sout[tid] + b2[tid];
}

// ---------------------------------------------------------------------------
// Encoder front linear: relu(res[1600,2] @ W.T[2,1024] + b) -> bf16 activations
// ---------------------------------------------------------------------------
__global__ void enc_front_k(const float* __restrict__ res, const float* __restrict__ W,
                            const float* __restrict__ bias,
                            __hip_bfloat16* __restrict__ out) {
  int idx = blockIdx.x * blockDim.x + threadIdx.x;
  if (idx >= BT_ * HID_) return;
  int m = idx / HID_, j = idx % HID_;
  float v = res[m * 2 + 0] * W[j * 2 + 0] + res[m * 2 + 1] * W[j * 2 + 1] + bias[j];
  out[idx] = __float2bfloat16(fmaxf(v, 0.f));
}

// ---------------------------------------------------------------------------
// WMMA GEMM: C[M,Nout] = X[M,K](bf16) @ W[Nout,K]^T(bf16) + bias (+ReLU)
// block = 128 threads = 4 waves; each wave computes 2 M-tiles x 4 N-tiles
// (8 accumulators): every B fragment feeds 2 WMMAs, every A fragment 4.
// Block tile 32 x 256. grid = (Nout/256, M/32)
// ---------------------------------------------------------------------------
template <bool RELU, bool OUT_BF16>
__global__ void gemm_xwt_bias(const __hip_bfloat16* __restrict__ Xb,
                              const __hip_bfloat16* __restrict__ Wb,
                              const float* __restrict__ bias,
                              void* __restrict__ Cout, int K, int Nout) {
  const __bf16* X = (const __bf16*)Xb;
  const __bf16* W = (const __bf16*)Wb;
  int lane = threadIdx.x & 31;
  int wave = threadIdx.x >> 5;                  // 0..3
  int mb  = blockIdx.y * 32;
  int nb0 = blockIdx.x * 256 + wave * 64;
  v8f acc[2][4] = {};
  for (int kk = 0; kk < K; kk += 32) {
    if (kk + 32 < K)
      __builtin_prefetch(W + (size_t)(nb0 + (lane & 15)) * K + kk + 32, 0, 1);
    v16bf a0 = frag_bf16(X, K, mb,      kk, lane);
    v16bf a1 = frag_bf16(X, K, mb + 16, kk, lane);
#pragma unroll
    for (int j = 0; j < 4; ++j) {
      v16bf b = frag_bf16(W, K, nb0 + 16 * j, kk, lane);
      acc[0][j] = wmma_bf16(a0, b, acc[0][j]);
      acc[1][j] = wmma_bf16(a1, b, acc[1][j]);
    }
  }
#pragma unroll
  for (int mt = 0; mt < 2; ++mt) {
    int rbase = mb + mt * 16 + ((lane >> 4) << 3);
#pragma unroll
    for (int j = 0; j < 4; ++j) {
      int col  = nb0 + 16 * j + (lane & 15);
      float bv = bias[col];
#pragma unroll
      for (int i = 0; i < 8; ++i) {
        float v = acc[mt][j][i] + bv;
        if (RELU) v = fmaxf(v, 0.f);
        if constexpr (OUT_BF16)
          ((__hip_bfloat16*)Cout)[(size_t)(rbase + i) * Nout + col] = __float2bfloat16(v);
        else
          ((float*)Cout)[(size_t)(rbase + i) * Nout + col] = v;
      }
    }
  }
}

// ---------------------------------------------------------------------------
// One GRU timestep. Hidden state kept in two forms: fp32 (exact carry) and
// bf16 (matmul operand). The bf16 state (32 x H) is staged into LDS with
// CDNA5 async global->LDS b128 copies; each wave owns one 16-column slice
// across all 3 gates and BOTH 16-row tiles (6 accumulators): every B (weight)
// fragment feeds 2 WMMAs, every A fragment 3; weights are fetched exactly
// once per block. grid = H/64, block = 128 (4 waves), dyn LDS = 32*H*2 bytes.
// ---------------------------------------------------------------------------
__global__ void gru_step_k(const float* __restrict__ h_in32,
                           const __hip_bfloat16* __restrict__ h_in16,
                           const __hip_bfloat16* __restrict__ Whhb,
                           const float* __restrict__ bhh,
                           const float* __restrict__ xi,       // [B*T, 3H] fp32
                           float* __restrict__ h_out32,
                           __hip_bfloat16* __restrict__ h_out16,
                           __hip_bfloat16* __restrict__ y,     // [B*T, H] bf16
                           int H, int t) {
  extern __shared__ char smem_raw[];
  __bf16* sH = (__bf16*)smem_raw;                 // [B_ x H] bf16
  const __bf16* W = (const __bf16*)Whhb;

  // ---- async stage h_in16 (32*H*2 bytes) into LDS ----
  {
    const int nbytes = B_ * H * 2;                // 2048B/iter @128 thr: no tail
    unsigned lds_base = (unsigned)(uintptr_t)(void*)sH;  // low 32b = LDS offset
    const char* gsrc = (const char*)h_in16;
    for (int off = threadIdx.x * 16; off < nbytes; off += blockDim.x * 16)
      async_g2lds_b128(lds_base + (unsigned)off, gsrc + off);
    wait_asynccnt0();
    __syncthreads();
  }

  int lane = threadIdx.x & 31;
  int wave = threadIdx.x >> 5;                    // 0..3
  int cb = blockIdx.x * 64 + wave * 16;
  v8f acc[2][3] = {};                             // [row-tile][gate r,z,n]
  for (int kk = 0; kk < H; kk += 32) {
    v16bf a0 = frag_bf16(sH, H, 0,  kk, lane);    // ds_load from LDS
    v16bf a1 = frag_bf16(sH, H, 16, kk, lane);
    v16bf bR = frag_bf16(W, H,          cb, kk, lane);
    v16bf bZ = frag_bf16(W, H,     H + cb, kk, lane);
    v16bf bN = frag_bf16(W, H, 2 * H + cb, kk, lane);
    acc[0][0] = wmma_bf16(a0, bR, acc[0][0]);
    acc[1][0] = wmma_bf16(a1, bR, acc[1][0]);
    acc[0][1] = wmma_bf16(a0, bZ, acc[0][1]);
    acc[1][1] = wmma_bf16(a1, bZ, acc[1][1]);
    acc[0][2] = wmma_bf16(a0, bN, acc[0][2]);
    acc[1][2] = wmma_bf16(a1, bN, acc[1][2]);
  }
  int col = cb + (lane & 15);
  float br = bhh[col], bz = bhh[H + col], bn = bhh[2 * H + col];
#pragma unroll
  for (int mt = 0; mt < 2; ++mt) {
    int rbase = mt * 16 + ((lane >> 4) << 3);
#pragma unroll
    for (int i = 0; i < 8; ++i) {
      int b = rbase + i;
      size_t mrow = (size_t)b * T_ + t;
      const float* xr = xi + mrow * (size_t)(3 * H);
      float ir = xr[col], iz = xr[H + col], in = xr[2 * H + col];
      float ho = h_in32[(size_t)b * H + col];
      float r  = sigmf(ir + acc[mt][0][i] + br);
      float z  = sigmf(iz + acc[mt][1][i] + bz);
      float n  = tanhf(in + r * (acc[mt][2][i] + bn));
      float hn = (1.f - z) * n + z * ho;
      h_out32[(size_t)b * H + col] = hn;
      __hip_bfloat16 hb = __float2bfloat16(hn);
      h_out16[(size_t)b * H + col] = hb;
      y[mrow * (size_t)H + col]    = hb;
    }
  }
}

// ---------------------------------------------------------------------------
// Final linear: out[1600,2] = d[1600,256](bf16) @ W.T[2,256] + b — tiny
// ---------------------------------------------------------------------------
__global__ void final_linear_k(const __hip_bfloat16* __restrict__ dI,
                               const float* __restrict__ W,
                               const float* __restrict__ b, float* __restrict__ out) {
  int idx = blockIdx.x * blockDim.x + threadIdx.x;
  if (idx >= BT_ * 2) return;
  int m = idx >> 1, c = idx & 1;
  const __hip_bfloat16* row = dI + (size_t)m * OUTH_;
  const float* w = W + (size_t)c * OUTH_;
  float s = b[c];
  for (int k = 0; k < OUTH_; ++k) s += __bfloat162float(row[k]) * w[k];
  out[idx] = s;
}

// ---------------------------------------------------------------------------
// Host orchestration
// ---------------------------------------------------------------------------
extern "C" void kernel_launch(void* const* d_in, const int* in_sizes, int n_in,
                              void* d_out, int out_size, void* d_ws, size_t ws_size,
                              hipStream_t stream) {
  const float* x        = (const float*)d_in[0];
  const int*   ei       = (const int*)  d_in[1];
  const float* gcn_W1   = (const float*)d_in[2];
  const float* gcn_b1   = (const float*)d_in[3];
  const float* gcn_W2   = (const float*)d_in[4];
  const float* gcn_b2   = (const float*)d_in[5];
  const float* enc_fl_W = (const float*)d_in[6];
  const float* enc_fl_b = (const float*)d_in[7];
  const float* enc_Wih  = (const float*)d_in[8];
  const float* enc_Whh  = (const float*)d_in[9];
  const float* enc_bih  = (const float*)d_in[10];
  const float* enc_bhh  = (const float*)d_in[11];
  const float* dec_fl_W = (const float*)d_in[12];
  const float* dec_fl_b = (const float*)d_in[13];
  const float* dec_Wih0 = (const float*)d_in[14];
  const float* dec_Whh0 = (const float*)d_in[15];
  const float* dec_bih0 = (const float*)d_in[16];
  const float* dec_bhh0 = (const float*)d_in[17];
  const float* dec_Wih1 = (const float*)d_in[18];
  const float* dec_Whh1 = (const float*)d_in[19];
  const float* dec_bih1 = (const float*)d_in[20];
  const float* dec_bhh1 = (const float*)d_in[21];
  const float* dec_linW = (const float*)d_in[22];
  const float* dec_linb = (const float*)d_in[23];

  // ---- workspace layout ----
  unsigned char* p = (unsigned char*)d_ws;
  auto take = [&](size_t bytes) {
    unsigned char* r = p;
    p += (bytes + 255) & ~(size_t)255;
    return (void*)r;
  };
  __hip_bfloat16* enc_Wih_b = (__hip_bfloat16*)take((size_t)2 * 3 * HID_ * HID_ * 2);
  __hip_bfloat16* enc_Whh_b = (__hip_bfloat16*)take((size_t)2 * 3 * HID_ * HID_ * 2);
  __hip_bfloat16* dec_flW_b = (__hip_bfloat16*)take((size_t)HID_ * HID_ * 2);
  __hip_bfloat16* dec_Wih0b = (__hip_bfloat16*)take((size_t)3 * OUTH_ * HID_ * 2);
  __hip_bfloat16* dec_Whh0b = (__hip_bfloat16*)take((size_t)3 * OUTH_ * OUTH_ * 2);
  __hip_bfloat16* dec_Wih1b = (__hip_bfloat16*)take((size_t)3 * OUTH_ * OUTH_ * 2);
  __hip_bfloat16* dec_Whh1b = (__hip_bfloat16*)take((size_t)3 * OUTH_ * OUTH_ * 2);
  float*          res    = (float*)take((size_t)BT_ * 2 * 4);
  __hip_bfloat16* act_a  = (__hip_bfloat16*)take((size_t)BT_ * HID_ * 2);
  __hip_bfloat16* act_b  = (__hip_bfloat16*)take((size_t)BT_ * HID_ * 2);
  float*          xi     = (float*)take((size_t)BT_ * 3 * HID_ * 4);
  float*          h32_0  = (float*)take((size_t)B_ * HID_ * 4);
  float*          h32_1  = (float*)take((size_t)B_ * HID_ * 4);
  __hip_bfloat16* h16_0  = (__hip_bfloat16*)take((size_t)B_ * HID_ * 2);
  __hip_bfloat16* h16_1  = (__hip_bfloat16*)take((size_t)B_ * HID_ * 2);

  auto cvt = [&](const float* s, __hip_bfloat16* d, long long n) {
    f32_to_bf16_k<<<(unsigned)((n + 255) / 256), 256, 0, stream>>>(s, d, n);
  };
  cvt(enc_Wih,  enc_Wih_b, (long long)2 * 3 * HID_ * HID_);
  cvt(enc_Whh,  enc_Whh_b, (long long)2 * 3 * HID_ * HID_);
  cvt(dec_fl_W, dec_flW_b, (long long)HID_ * HID_);
  cvt(dec_Wih0, dec_Wih0b, (long long)3 * OUTH_ * HID_);
  cvt(dec_Whh0, dec_Whh0b, (long long)3 * OUTH_ * OUTH_);
  cvt(dec_Wih1, dec_Wih1b, (long long)3 * OUTH_ * OUTH_);
  cvt(dec_Whh1, dec_Whh1b, (long long)3 * OUTH_ * OUTH_);

  // ---- GCN front-end ----
  gcn_kernel<<<BT_, 128, 0, stream>>>(x, ei, gcn_W1, gcn_b1, gcn_W2, gcn_b2, res);

  // ---- Encoder front linear + ReLU ----
  enc_front_k<<<(BT_ * HID_ + 255) / 256, 256, 0, stream>>>(res, enc_fl_W, enc_fl_b, act_a);

  auto run_gru = [&](const __hip_bfloat16* Wih_b, const float* bih,
                     const __hip_bfloat16* Whh_b, const float* bhh,
                     __hip_bfloat16* x_in, __hip_bfloat16* y_out, int K, int H) {
    gemm_xwt_bias<false, false><<<dim3(3 * H / 256, BT_ / 32), 128, 0, stream>>>(
        x_in, Wih_b, bih, xi, K, 3 * H);
    zero_f32_k<<<(B_ * H + 255) / 256, 256, 0, stream>>>(h32_0, B_ * H);
    zero_b16_k<<<(B_ * H + 255) / 256, 256, 0, stream>>>(h16_0, B_ * H);
    size_t lds = (size_t)B_ * H * 2;
    for (int t = 0; t < T_; ++t) {
      float*          hin32  = (t & 1) ? h32_1 : h32_0;
      float*          hout32 = (t & 1) ? h32_0 : h32_1;
      __hip_bfloat16* hin16  = (t & 1) ? h16_1 : h16_0;
      __hip_bfloat16* hout16 = (t & 1) ? h16_0 : h16_1;
      gru_step_k<<<H / 64, 128, lds, stream>>>(hin32, hin16, Whh_b, bhh, xi,
                                               hout32, hout16, y_out, H, t);
    }
  };

  // ---- Encoder GRU: 2 layers, H = 1024 ----
  for (int l = 0; l < 2; ++l) {
    run_gru(enc_Wih_b + (size_t)l * 3 * HID_ * HID_, enc_bih + (size_t)l * 3 * HID_,
            enc_Whh_b + (size_t)l * 3 * HID_ * HID_, enc_bhh + (size_t)l * 3 * HID_,
            act_a, act_b, HID_, HID_);
    __hip_bfloat16* tmp = act_a; act_a = act_b; act_b = tmp;
  }

  // ---- Decoder front linear + ReLU (1024 -> 1024, WMMA, bf16 out) ----
  gemm_xwt_bias<true, true><<<dim3(HID_ / 256, BT_ / 32), 128, 0, stream>>>(
      act_a, dec_flW_b, dec_fl_b, act_b, HID_, HID_);
  { __hip_bfloat16* tmp = act_a; act_a = act_b; act_b = tmp; }

  // ---- Decoder GRU layer 0: input 1024, H = 256 ----
  run_gru(dec_Wih0b, dec_bih0, dec_Whh0b, dec_bhh0, act_a, act_b, HID_, OUTH_);
  { __hip_bfloat16* tmp = act_a; act_a = act_b; act_b = tmp; }

  // ---- Decoder GRU layer 1: input 256, H = 256 ----
  run_gru(dec_Wih1b, dec_bih1, dec_Whh1b, dec_bhh1, act_a, act_b, OUTH_, OUTH_);
  { __hip_bfloat16* tmp = act_a; act_a = act_b; act_b = tmp; }

  // ---- Final linear (256 -> 2) ----
  final_linear_k<<<(BT_ * 2 + 255) / 256, 256, 0, stream>>>(act_a, dec_linW, dec_linb,
                                                            (float*)d_out);
}
